// EdgeDecoder_13142599925847
// MI455X (gfx1250) — matrix operand order repaired and checked by reading, not compiled, not run on hardware
//
#include <hip/hip_runtime.h>

typedef __attribute__((ext_vector_type(16))) __bf16 v16bf;
typedef __attribute__((ext_vector_type(8)))  __bf16 v8bf;
typedef __attribute__((ext_vector_type(4)))  __bf16 v4bf;
typedef __attribute__((ext_vector_type(2)))  __bf16 v2bf;
typedef __attribute__((ext_vector_type(8)))  float  v8f;
typedef __attribute__((ext_vector_type(4)))  float  v4f;

#define NW      6      // waves per block (192 threads): 155.9 KB LDS -> 2 blocks/WGP (3 waves/SIMD)
#define H       128
#define W1_LD   264    // padded LDS row stride (bf16) for 256-wide rows: 528B, 16B aligned, bank-skewed
#define W2_LD   136    // padded LDS row stride (bf16) for 128-wide rows: 272B
#define ACT_LD  136

// Scheduling fence: VALU|SALU|VMEM|trans|non-mem may cross; DS and WMMA may NOT.
// Caps the DS prefetch window at program-order depth (2 fragments) -> <256 VGPRs.
#define SCHED_FENCE() __builtin_amdgcn_sched_barrier(0x0477)

__device__ __forceinline__ v16bf cvt16(v4f f0, v4f f1, v4f f2, v4f f3) {
  v16bf r;
#pragma unroll
  for (int i = 0; i < 4; ++i) {
    r[i]      = (__bf16)f0[i];
    r[4 + i]  = (__bf16)f1[i];
    r[8 + i]  = (__bf16)f2[i];
    r[12 + i] = (__bf16)f3[i];
  }
  return r;
}

__device__ __forceinline__ v16bf combine16(v8bf lo, v8bf hi) {
  v16bf r;
#pragma unroll
  for (int i = 0; i < 8; ++i) { r[i] = lo[i]; r[8 + i] = hi[i]; }
  return r;
}

__device__ __forceinline__ v16bf lds_frag(const __bf16* p) {
  // 32 contiguous bytes -> two ds_load_b128
  return combine16(*(const v8bf*)p, *(const v8bf*)(p + 8));
}

// Layer-1 A fragment from the f32 gather (fallback path).
// k order used here for A and B: lane(m,half); elem e <-> K=16*half+e
__device__ __forceinline__ v16bf gather_frag_f32(const float* ubase, const float* rbase,
                                                 int s, int half) {
  const int k0 = s * 32;
  const float* src = (s < 4) ? (ubase + k0) : (rbase + (k0 - H));
  v4f f0 = *(const v4f*)(src + 16 * half);
  v4f f1 = *(const v4f*)(src + 16 * half + 4);
  v4f f2 = *(const v4f*)(src + 16 * half + 8);
  v4f f3 = *(const v4f*)(src + 16 * half + 12);
  return cvt16(f0, f1, f2, f3);
}

// Layer-1 A fragment straight from pre-converted bf16 tables: 2 x global_load_b128, no cvt.
__device__ __forceinline__ v16bf gather_frag_bf16(const __bf16* ubase, const __bf16* rbase,
                                                  int s, int half) {
  const int k0 = s * 32;
  const __bf16* src = (s < 4) ? (ubase + k0) : (rbase + (k0 - H));
  return combine16(*(const v8bf*)(src + 16 * half), *(const v8bf*)(src + 16 * half + 8));
}

// A fragment from an LDS activation pad [16][ACT_LD]
__device__ __forceinline__ v16bf act_frag(const __bf16* actw, int m, int half, int s) {
  const __bf16* p = actw + m * ACT_LD + s * 32 + 16 * half;
  return combine16(*(const v8bf*)p, *(const v8bf*)(p + 8));
}

__device__ __forceinline__ v8f splat8(float x) {
  v8f c;
#pragma unroll
  for (int i = 0; i < 8; ++i) c[i] = x;
  return c;
}

// transpose-store one D tile (bias already in acc): cvt_pk then packed bf16 ReLU
__device__ __forceinline__ void store_tile(__bf16* actw, int m, int half, int t, v8f c) {
#pragma unroll
  for (int r = 0; r < 8; r += 2) {
    v2bf pk;                          // v_cvt_pk_bf16_f32
    pk[0] = (__bf16)c[r];
    pk[1] = (__bf16)c[r + 1];
    unsigned u = __builtin_bit_cast(unsigned, pk);
    asm("v_pk_max_num_bf16 %0, %1, 0" : "=v"(u) : "v"(u));   // packed ReLU
    v2bf mp = __builtin_bit_cast(v2bf, u);
    actw[(r     + 8 * half) * ACT_LD + t * 16 + m] = mp[0];
    actw[(r + 1 + 8 * half) * ACT_LD + t * 16 + m] = mp[1];
  }
}

// one-shot f32 -> bf16 table conversion into workspace
__global__ void cvt_table_kernel(const float* __restrict__ src, __bf16* __restrict__ dst,
                                 long n4) {                    // n4 = count/4
  long i = (long)blockIdx.x * blockDim.x + threadIdx.x;
  const long stride = (long)gridDim.x * blockDim.x;
  for (; i < n4; i += stride) {
    v4f f = *(const v4f*)(src + i * 4);
    v4bf b;
#pragma unroll
    for (int j = 0; j < 4; ++j) b[j] = (__bf16)f[j];
    *(v4bf*)(dst + i * 4) = b;
  }
}

template <int BF16G>
__global__ __launch_bounds__(NW * 32)
void edge_mlp_kernel(const float* __restrict__ zu, const float* __restrict__ zr,
                     const __bf16* __restrict__ zub, const __bf16* __restrict__ zrb,
                     const int* __restrict__ row, const int* __restrict__ col,
                     const float* __restrict__ W1, const float* __restrict__ B1,
                     const float* __restrict__ W2, const float* __restrict__ B2,
                     const float* __restrict__ W3, const float* __restrict__ B3,
                     float* __restrict__ out, int E, int numPairs) {
  __shared__ __bf16 w1s[H * W1_LD];          // 67,584 B
  __shared__ __bf16 w2s[H * W2_LD];          // 34,816 B
  __shared__ __bf16 w3b[H];                  // 256 B
  __shared__ float  b1s[H];
  __shared__ float  b2s[H];
  __shared__ __bf16 acts[NW * 32 * ACT_LD];  // 52,224 B   (total 155.9 KB -> 2 blocks/WGP)

  const int tid = threadIdx.x;

  // ---- stage weights to LDS as bf16 (once per block) ----
  for (int c = tid; c < H * 64; c += NW * 32) {          // W1: 128 x 256, float4 chunks
    int r = c >> 6, kc = (c & 63) << 2;
    v4f f = *(const v4f*)(W1 + r * 256 + kc);
    v4bf b;
#pragma unroll
    for (int i = 0; i < 4; ++i) b[i] = (__bf16)f[i];
    *(v4bf*)(w1s + r * W1_LD + kc) = b;
  }
  for (int c = tid; c < H * 32; c += NW * 32) {          // W2: 128 x 128
    int r = c >> 5, kc = (c & 31) << 2;
    v4f f = *(const v4f*)(W2 + r * H + kc);
    v4bf b;
#pragma unroll
    for (int i = 0; i < 4; ++i) b[i] = (__bf16)f[i];
    *(v4bf*)(w2s + r * W2_LD + kc) = b;
  }
  if (tid < H) { w3b[tid] = (__bf16)W3[tid]; b1s[tid] = B1[tid]; b2s[tid] = B2[tid]; }
  __syncthreads();

  const int lane = tid & 31;
  const int wave = tid >> 5;
  const int m    = lane & 15;   // A row index, or B/D column index
  const int half = lane >> 4;
  __bf16* actA = acts + wave * 32 * ACT_LD;
  __bf16* actB = actA + 16 * ACT_LD;
  const float b3v = B3[0];

  v16bf zf;
#pragma unroll
  for (int i = 0; i < 16; ++i) zf[i] = (__bf16)0.f;

  for (int pair = blockIdx.x * NW + wave; pair < numPairs; pair += gridDim.x * NW) {
    const int eBase = pair << 5;                  // 32 edges per wave-iteration
    int eA = eBase + m;      if (eA >= E) eA = E - 1;
    int eB = eBase + 16 + m; if (eB >= E) eB = E - 1;
    const long uoA = (long)row[eA] * H, roA = (long)col[eA] * H;
    const long uoB = (long)row[eB] * H, roB = (long)col[eB] * H;

    // ================= layer 1: [32x256] @ [256x128] =================
    v8f cA[8], cB[8];
#pragma unroll
    for (int t = 0; t < 8; ++t) {                 // bias folded into accumulator init
      float b = b1s[t * 16 + m];
      cA[t] = splat8(b);
      cB[t] = splat8(b);
    }
    {
      v16bf fA, fB;
      if (BF16G) {
        fA = gather_frag_bf16(zub + uoA, zrb + roA, 0, half);
        fB = gather_frag_bf16(zub + uoB, zrb + roB, 0, half);
      } else {
        fA = gather_frag_f32(zu + uoA, zr + roA, 0, half);
        fB = gather_frag_f32(zu + uoB, zr + roB, 0, half);
      }
      v16bf nfA = fA, nfB = fB;
      v16bf wb = lds_frag(w1s + (0 * 16 + m) * W1_LD + 0 * 32 + 16 * half);
#pragma unroll
      for (int i = 0; i < 64; ++i) {              // linearized (s,t); rotating prefetch
        const int s = i >> 3, t = i & 7;
        v16bf wbn;
        if (i + 1 < 64) {                          // B frag for step i+1, issued pre-WMMA
          const int s1 = (i + 1) >> 3, t1 = (i + 1) & 7;
          wbn = lds_frag(w1s + (t1 * 16 + m) * W1_LD + s1 * 32 + 16 * half);
        }
        if (t == 0 && s + 1 < 8) {                 // next-s gather, 8 WMMA-pairs of cover
          if (BF16G) {
            nfA = gather_frag_bf16(zub + uoA, zrb + roA, s + 1, half);
            nfB = gather_frag_bf16(zub + uoB, zrb + roB, s + 1, half);
          } else {
            nfA = gather_frag_f32(zu + uoA, zr + roA, s + 1, half);
            nfB = gather_frag_f32(zu + uoB, zr + roB, s + 1, half);
          }
        }
        cA[t] = __builtin_amdgcn_wmma_f32_16x16x32_bf16(false, fA, false, wb,
                                                        (short)0, cA[t], false, false);
        cB[t] = __builtin_amdgcn_wmma_f32_16x16x32_bf16(false, fB, false, wb,
                                                        (short)0, cB[t], false, false);
        if (i + 1 < 64) wb = wbn;
        if (t == 7 && s + 1 < 8) { fA = nfA; fB = nfB; }
        SCHED_FENCE();
      }
    }
#pragma unroll
    for (int t = 0; t < 8; ++t) {                 // D->A transpose through LDS
      store_tile(actA, m, half, t, cA[t]);
      store_tile(actB, m, half, t, cB[t]);
    }
    asm volatile("s_wait_dscnt 0x0" ::: "memory");   // wave-private LDS RAW

    // ================= layer 2: [32x128] @ [128x128] =================
#pragma unroll
    for (int t = 0; t < 8; ++t) {
      float b = b2s[t * 16 + m];
      cA[t] = splat8(b);
      cB[t] = splat8(b);
    }
    {
      v16bf fA = act_frag(actA, m, half, 0);
      v16bf fB = act_frag(actB, m, half, 0);
      v16bf nfA = fA, nfB = fB;
      v16bf wb = lds_frag(w2s + (0 * 16 + m) * W2_LD + 0 * 32 + 16 * half);
#pragma unroll
      for (int i = 0; i < 32; ++i) {
        const int s = i >> 3, t = i & 7;
        v16bf wbn;
        if (i + 1 < 32) {
          const int s1 = (i + 1) >> 3, t1 = (i + 1) & 7;
          wbn = lds_frag(w2s + (t1 * 16 + m) * W2_LD + s1 * 32 + 16 * half);
        }
        if (t == 0 && s + 1 < 4) {
          nfA = act_frag(actA, m, half, s + 1);
          nfB = act_frag(actB, m, half, s + 1);
        }
        cA[t] = __builtin_amdgcn_wmma_f32_16x16x32_bf16(false, fA, false, wb,
                                                        (short)0, cA[t], false, false);
        cB[t] = __builtin_amdgcn_wmma_f32_16x16x32_bf16(false, fB, false, wb,
                                                        (short)0, cB[t], false, false);
        if (i + 1 < 32) wb = wbn;
        if (t == 7 && s + 1 < 4) { fA = nfA; fB = nfB; }
        SCHED_FENCE();
      }
    }
#pragma unroll
    for (int t = 0; t < 8; ++t) {                 // all LDS reads above done; safe overwrite
      store_tile(actA, m, half, t, cA[t]);
      store_tile(actB, m, half, t, cB[t]);
    }
    asm volatile("s_wait_dscnt 0x0" ::: "memory");

    // ================= layer 3: [32x128] @ [128x1] via WMMA ==========
    // B fragment: W3 in column n=0 only; other columns zero.
    v8f dA = splat8((m == 0) ? b3v : 0.f);
    v8f dB = dA;
    {
      v16bf fA = act_frag(actA, m, half, 0);
      v16bf fB = act_frag(actB, m, half, 0);
      v16bf w3f = zf;
      if (m == 0) w3f = lds_frag(w3b + 16 * half);
#pragma unroll
      for (int s = 0; s < 4; ++s) {
        v16bf nfA = fA, nfB = fB, nw = w3f;
        if (s + 1 < 4) {
          nfA = act_frag(actA, m, half, s + 1);
          nfB = act_frag(actB, m, half, s + 1);
          nw = zf;
          if (m == 0) nw = lds_frag(w3b + (s + 1) * 32 + 16 * half);
        }
        dA = __builtin_amdgcn_wmma_f32_16x16x32_bf16(false, fA, false, w3f,
                                                     (short)0, dA, false, false);
        dB = __builtin_amdgcn_wmma_f32_16x16x32_bf16(false, fB, false, w3f,
                                                     (short)0, dB, false, false);
        fA = nfA; fB = nfB; w3f = nw;
        SCHED_FENCE();
      }
    }
    if (m == 0) {                                 // lanes 0 and 16 hold column 0
#pragma unroll
      for (int r = 0; r < 8; ++r) {
        int e0 = eBase + r + 8 * half;
        if (e0 < E) out[e0] = dA[r];
        int e1 = eBase + 16 + r + 8 * half;
        if (e1 < E) out[e1] = dB[r];
      }
    }
  }
}

extern "C" void kernel_launch(void* const* d_in, const int* in_sizes, int n_in,
                              void* d_out, int out_size, void* d_ws, size_t ws_size,
                              hipStream_t stream) {
  const float* zu = (const float*)d_in[0];
  const float* zr = (const float*)d_in[1];
  const int*  row = (const int*)d_in[2];
  const int*  col = (const int*)d_in[3];
  const float* W1 = (const float*)d_in[4];
  const float* B1 = (const float*)d_in[5];
  const float* W2 = (const float*)d_in[6];
  const float* B2 = (const float*)d_in[7];
  const float* W3 = (const float*)d_in[8];
  const float* B3 = (const float*)d_in[9];
  float* out = (float*)d_out;

  const int E = in_sizes[2];
  const int numTiles = (E + 15) / 16;
  const int numPairs = (numTiles + 1) / 2;

  // Balanced persistent grid: equalize iterations per wave, cap at 1024 blocks.
  const int MAXB = 1024;
  int iters = (numPairs + NW * MAXB - 1) / (NW * MAXB);
  if (iters < 1) iters = 1;
  int blocks = (numPairs + NW * iters - 1) / (NW * iters);
  if (blocks < 1) blocks = 1;

  const size_t nU = (size_t)in_sizes[0], nR = (size_t)in_sizes[1];
  const size_t need = (nU + nR) * sizeof(__bf16);

  if (ws_size >= need) {
    __bf16* zub = (__bf16*)d_ws;
    __bf16* zrb = zub + nU;
    cvt_table_kernel<<<2048, 256, 0, stream>>>(zu, zub, (long)(nU / 4));
    cvt_table_kernel<<<1024, 256, 0, stream>>>(zr, zrb, (long)(nR / 4));
    edge_mlp_kernel<1><<<blocks, NW * 32, 0, stream>>>(zu, zr, zub, zrb, row, col,
                                                       W1, B1, W2, B2, W3, B3,
                                                       out, E, numPairs);
  } else {
    edge_mlp_kernel<0><<<blocks, NW * 32, 0, stream>>>(zu, zr, nullptr, nullptr, row, col,
                                                       W1, B1, W2, B2, W3, B3,
                                                       out, E, numPairs);
  }
}
